// LSTMPositionPredictor_67465346285956
// MI455X (gfx1250) — matrix operand order repaired and checked by reading, not compiled, not run on hardware
//
#include <hip/hip_runtime.h>
#include <hip/hip_bf16.h>

// Problem constants (match reference globals)
#define B_   1024
#define S_   64
#define T_   32
#define H_   256
#define IN_  4
#define GH_  (4 * H_)          // 1024 gate outputs per direction

// LDS row strides (padded to avoid bank conflicts; 64 banks x 4B)
#define HPAD   264             // ushort stride for bf16 h rows (16B-mult, +4 dword skew/row)
#define H32PAD 260             // f32 stride for decoder f32 h copy

typedef __attribute__((ext_vector_type(16))) __bf16        v16bf;
typedef __attribute__((ext_vector_type(8)))  float         v8f;
typedef __attribute__((ext_vector_type(4)))  unsigned int  v4u;

union V16U { v4u q[2]; v16bf v; };

__device__ __host__ __forceinline__ unsigned short f2bf(float f) {
  union { float f; unsigned u; } c; c.f = f;
  unsigned u = c.u + 0x7FFFu + ((c.u >> 16) & 1u);   // round-to-nearest-even
  return (unsigned short)(u >> 16);
}

__device__ __forceinline__ float sigf(float x) { return 1.0f / (1.0f + __expf(-x)); }
__device__ __forceinline__ float tanh_f(float x) {
  float xc = fminf(fmaxf(x, -15.0f), 15.0f);
  float e = __expf(-2.0f * xc);
  return (1.0f - e) / (1.0f + e);
}

// ---- WMMA operand loaders -------------------------------------------------
// A (16x32 bf16, M=batch rows, K=h cols), from LDS h buffer.
// ISA layout: lane m (0..15): K=0..7 (v0..3), K=16..23 (v4..7); lane m+16: K=8..15, K=24..31.
__device__ __forceinline__ v16bf load_A(const unsigned short* hbuf, int lane, int kc) {
  const int row = lane & 15;
  const int kk  = kc * 32 + ((lane >> 4) << 3);
  V16U u;
  u.q[0] = *(const v4u*)(hbuf + row * HPAD + kk);
  u.q[1] = *(const v4u*)(hbuf + row * HPAD + kk + 16);
  return u.v;
}

// B (32x16 bf16) from pre-swizzled global weights: operand block (nt,kc) is a
// contiguous 1KB run; lane l reads 32B at +l*32 -> fully coalesced b128 pairs.
__device__ __forceinline__ v16bf load_B(const unsigned short* __restrict__ wswz,
                                        int lane, int nt, int kc) {
  const unsigned short* p = wswz + ((((nt << 3) + kc) << 9) + (lane << 4));
  V16U u;
  u.q[0] = *(const v4u*)(p);
  u.q[1] = *(const v4u*)(p + 8);
  return u.v;
}

// ---- One LSTM step for a 16-row batch tile --------------------------------
// Gates G(16x1024) = h(16x256) @ wh.T + x(16x4) @ wi.T + b, then elementwise.
// Wave w owns h-columns [32w,32w+32): computes gate tiles nt = g*16 + 2w + t.
template<bool WF32>
__device__ __forceinline__ void lstm_step(
    const unsigned short* __restrict__ wswz,   // swizzled bf16 wh (4H x H)
    const float* __restrict__ wi,              // (4H x 4) f32
    const float* __restrict__ bias,            // (4H)    f32
    const float* __restrict__ xt,              // LDS [16][4] f32
    unsigned short* hbuf,                      // LDS [16][HPAD] bf16 (in/out)
    float* h32,                                // LDS [16][H32PAD] f32 (out, decoder)
    float* creg,                               // 16 regs: cell state (in/out)
    int lane, int wave)
{
  const int half = lane >> 4;

  // Accumulator init: bias + x @ wi.T (K=4 input term in VALU)
  v8f acc[4][2];
#pragma unroll
  for (int g = 0; g < 4; ++g) {
#pragma unroll
    for (int t = 0; t < 2; ++t) {
      const int n = g * H_ + 32 * wave + 16 * t + (lane & 15);
      const float4 wv = *(const float4*)(wi + n * IN_);
      const float  bb = bias[n];
#pragma unroll
      for (int r = 0; r < 8; ++r) {
        const float4 xr = *(const float4*)(xt + (r + 8 * half) * IN_);
        acc[g][t][r] = bb + xr.x * wv.x + xr.y * wv.y + xr.z * wv.z + xr.w * wv.w;
      }
    }
  }

  // Matmul: kc is a REAL loop (8 iters) to bound register live ranges.
  // Per iteration: 1 A operand (LDS), 8 coalesced B operands (global/L2),
  // then 8 independent WMMAs (one per gate tile -> ILP inside the iteration,
  // no cross-iteration hoisting, no spills).
#pragma unroll 1
  for (int kc = 0; kc < 8; ++kc) {
    const v16bf Aop = load_A(hbuf, lane, kc);
    v16bf Bv[8];
#pragma unroll
    for (int u = 0; u < 8; ++u) {                 // u = g*2 + t
      const int nt = ((u >> 1) << 4) + (wave << 1) + (u & 1);
      Bv[u] = load_B(wswz, lane, nt, kc);
    }
#pragma unroll
    for (int u = 0; u < 8; ++u) {
      acc[u >> 1][u & 1] = __builtin_amdgcn_wmma_f32_16x16x32_bf16(
          false, Aop, false, Bv[u], (short)0, acc[u >> 1][u & 1], false, false);
    }
  }

  __syncthreads();   // everyone done reading old h before we overwrite (WAR)

  // Elementwise LSTM update entirely in WMMA C-layout: VGPR r, lane l ->
  // (M = r + 8*(l>>4), col = 32w + 16t + (l&15)); gates i/f/g/o share position.
#pragma unroll
  for (int t = 0; t < 2; ++t) {
#pragma unroll
    for (int r = 0; r < 8; ++r) {
      const int M = r + 8 * half;
      const int j = 32 * wave + 16 * t + (lane & 15);
      float iv = acc[0][t][r], fv = acc[1][t][r];
      float gv = acc[2][t][r], ov = acc[3][t][r];
      float c2 = sigf(fv) * creg[t * 8 + r] + sigf(iv) * tanh_f(gv);
      float h2 = sigf(ov) * tanh_f(c2);
      creg[t * 8 + r] = c2;
      hbuf[M * HPAD + j] = f2bf(h2);
      if (WF32) h32[M * H32PAD + j] = h2;
    }
  }
}

// ---- Weight conversion: fp32 (4H x H) -> bf16 in WMMA B-operand order -----
// dst[(nt*8+kc)*512 + lane*16 + j] = bf16(wh[nt*16 + (lane&15)][kc*32 + (lane>>4)*8 + (j&7) + (j>>3)*16])
__global__ void convert_wh_kernel(const float* __restrict__ src,
                                  unsigned short* __restrict__ dst) {
  int i    = blockIdx.x * blockDim.x + threadIdx.x;  // 0..262143
  int j    = i & 15;
  int lane = (i >> 4) & 31;
  int blk  = i >> 9;
  int kc   = blk & 7;
  int nt   = blk >> 3;
  int n    = (nt << 4) + (lane & 15);
  int k    = (kc << 5) + ((lane >> 4) << 3) + (j & 7) + ((j >> 3) << 4);
  dst[i] = f2bf(src[n * H_ + k]);
}

// ---- Encoder: grid (64 batch tiles, 2 directions) -------------------------
__global__ __launch_bounds__(256) void encoder_kernel(
    const float* __restrict__ x,
    const float* __restrict__ wi_f, const float* __restrict__ b_f,
    const float* __restrict__ wi_b, const float* __restrict__ b_b,
    const unsigned short* __restrict__ whf_swz,
    const unsigned short* __restrict__ whb_swz,
    unsigned short* __restrict__ h_ws,   // [2][B][H] bf16 final h
    float* __restrict__ c_ws)            // [2][B][H] f32  final c
{
  __shared__ __align__(16) unsigned short hbuf[16 * HPAD];
  __shared__ __align__(16) float xt[16 * IN_];

  const int tid  = threadIdx.x;
  const int lane = tid & 31;
  const int wave = tid >> 5;
  const int b0   = blockIdx.x * 16;
  const int dir  = blockIdx.y;                 // 0 = forward, 1 = backward

  const float* wi = dir ? wi_b : wi_f;
  const float* bs = dir ? b_b  : b_f;
  const unsigned short* wswz = dir ? whb_swz : whf_swz;

  float creg[16];
#pragma unroll
  for (int i = 0; i < 16; ++i) creg[i] = 0.0f;
  for (int i = tid; i < 16 * HPAD; i += 256) hbuf[i] = 0;
  __syncthreads();

  for (int s = 0; s < S_; ++s) {
    const int sidx = dir ? (S_ - 1 - s) : s;
    if (tid < 64) {
      int row = tid >> 2, k = tid & 3;
      xt[tid] = x[((size_t)(b0 + row) * S_ + sidx) * IN_ + k];
    }
    __syncthreads();                            // xt + previous h visible
    lstm_step<false>(wswz, wi, bs, xt, hbuf, nullptr, creg, lane, wave);
  }
  __syncthreads();

  // Hand off final h (bf16) and c (f32) to the decoder via workspace
  const size_t dbase = (size_t)dir * B_ * H_;
  for (int i = tid; i < 16 * H_; i += 256) {
    int row = i >> 8, col = i & 255;
    h_ws[dbase + (size_t)(b0 + row) * H_ + col] = hbuf[row * HPAD + col];
  }
  const int half = lane >> 4;
#pragma unroll
  for (int t = 0; t < 2; ++t) {
#pragma unroll
    for (int r = 0; r < 8; ++r) {
      int M = r + 8 * half;
      int col = 32 * wave + 16 * t + (lane & 15);
      c_ws[dbase + (size_t)(b0 + M) * H_ + col] = creg[t * 8 + r];
    }
  }
}

// ---- Decoder: grid 64 batch tiles; both directions + projection per step --
__global__ __launch_bounds__(256) void decoder_kernel(
    const float* __restrict__ x, const int* __restrict__ lengths,
    const float* __restrict__ dwi_f, const float* __restrict__ db_f,
    const float* __restrict__ dwi_b, const float* __restrict__ db_b,
    const float* __restrict__ gw, const float* __restrict__ gb,
    const unsigned short* __restrict__ dwhf_swz,
    const unsigned short* __restrict__ dwhb_swz,
    const unsigned short* __restrict__ h_ws, const float* __restrict__ c_ws,
    float* __restrict__ out)
{
  __shared__ __align__(16) unsigned short h_f[16 * HPAD];
  __shared__ __align__(16) unsigned short h_b[16 * HPAD];
  __shared__ __align__(16) float h32f[16 * H32PAD];
  __shared__ __align__(16) float h32b[16 * H32PAD];
  __shared__ __align__(16) float xt[16 * IN_];

  const int tid  = threadIdx.x;
  const int lane = tid & 31;
  const int wave = tid >> 5;
  const int half = lane >> 4;
  const int b0   = blockIdx.x * 16;

  // Stage encoder-final h into LDS (bf16, padded rows)
  for (int i = tid; i < 16 * H_; i += 256) {
    int row = i >> 8, col = i & 255;
    h_f[row * HPAD + col] = h_ws[(size_t)(b0 + row) * H_ + col];
    h_b[row * HPAD + col] = h_ws[(size_t)B_ * H_ + (size_t)(b0 + row) * H_ + col];
  }
  // Cell state into registers (same (lane,reg) mapping as encoder)
  float cf[16], cb[16];
#pragma unroll
  for (int t = 0; t < 2; ++t) {
#pragma unroll
    for (int r = 0; r < 8; ++r) {
      int M = r + 8 * half;
      int col = 32 * wave + 16 * t + (lane & 15);
      cf[t * 8 + r] = c_ws[(size_t)(b0 + M) * H_ + col];
      cb[t * 8 + r] = c_ws[(size_t)B_ * H_ + (size_t)(b0 + M) * H_ + col];
    }
  }
  // x0 = x[b, lengths[b]-1, :]
  if (tid < 64) {
    int row = tid >> 2, k = tid & 3;
    int len = lengths[b0 + row];
    xt[tid] = x[((size_t)(b0 + row) * S_ + (len - 1)) * IN_ + k];
  }
  __syncthreads();

  const long BT4 = (long)B_ * T_ * IN_;
  for (int t = 0; t < T_; ++t) {
    lstm_step<true>(dwhf_swz, dwi_f, db_f, xt, h_f, h32f, cf, lane, wave);
    lstm_step<true>(dwhb_swz, dwi_b, db_b, xt, h_b, h32b, cb, lane, wave);
    __syncthreads();   // h32f/h32b complete for projection

    // g(16x8) = [hf2 hb2](16x512) @ gw.T + gb   (small: plain VALU)
    if (tid < 128) {
      int row = tid >> 3, cj = tid & 7;
      const float* gwrow = gw + cj * (2 * H_);
      float a = gb[cj];
      for (int k = 0; k < H_; ++k) a += h32f[row * H32PAD + k] * gwrow[k];
      for (int k = 0; k < H_; ++k) a += h32b[row * H32PAD + k] * gwrow[H_ + k];
      long base = ((long)(b0 + row) * T_ + t) * IN_;
      if (cj < 4) {
        out[base + cj] = a;                    // predictions (mu)
        xt[row * IN_ + cj] = a;                // next-step input
      } else {
        out[BT4 + base + (cj - 4)] = __expf(a);// sigma
        if (cj == 6 && t == T_ - 1)
          out[2 * BT4 + (b0 + row)] = tanh_f(a); // rho (last step)
      }
    }
    __syncthreads();   // xt published; h32 read-before-next-write ordered
  }
}

extern "C" void kernel_launch(void* const* d_in, const int* in_sizes, int n_in,
                              void* d_out, int out_size, void* d_ws, size_t ws_size,
                              hipStream_t stream) {
  const float* x        = (const float*)d_in[0];
  const int*   lengths  = (const int*)d_in[1];
  // d_in[2] = target_len (compile-time T_=32)
  const float* enc_wi_f = (const float*)d_in[3];
  const float* enc_wh_f = (const float*)d_in[4];
  const float* enc_b_f  = (const float*)d_in[5];
  const float* enc_wi_b = (const float*)d_in[6];
  const float* enc_wh_b = (const float*)d_in[7];
  const float* enc_b_b  = (const float*)d_in[8];
  const float* dec_wi_f = (const float*)d_in[9];
  const float* dec_wh_f = (const float*)d_in[10];
  const float* dec_b_f  = (const float*)d_in[11];
  const float* dec_wi_b = (const float*)d_in[12];
  const float* dec_wh_b = (const float*)d_in[13];
  const float* dec_b_b  = (const float*)d_in[14];
  const float* gw       = (const float*)d_in[15];
  const float* gb       = (const float*)d_in[16];

  const size_t WN = (size_t)GH_ * H_;            // 262144 elems per wh matrix
  unsigned short* wsu = (unsigned short*)d_ws;
  unsigned short* ef  = wsu;                      // swizzled bf16 weights (2 MB total)
  unsigned short* eb  = wsu + WN;
  unsigned short* df  = wsu + 2 * WN;
  unsigned short* db  = wsu + 3 * WN;
  unsigned short* h_ws = wsu + 4 * WN;            // [2][B][H] bf16 (1 MB)
  float* c_ws = (float*)(wsu + 4 * WN + 2 * (size_t)B_ * H_); // [2][B][H] f32 (2 MB)

  dim3 cgrid((unsigned)(WN / 256));
  convert_wh_kernel<<<cgrid, 256, 0, stream>>>(enc_wh_f, ef);
  convert_wh_kernel<<<cgrid, 256, 0, stream>>>(enc_wh_b, eb);
  convert_wh_kernel<<<cgrid, 256, 0, stream>>>(dec_wh_f, df);
  convert_wh_kernel<<<cgrid, 256, 0, stream>>>(dec_wh_b, db);

  encoder_kernel<<<dim3(B_ / 16, 2), 256, 0, stream>>>(
      x, enc_wi_f, enc_b_f, enc_wi_b, enc_b_b, ef, eb, h_ws, c_ws);

  decoder_kernel<<<dim3(B_ / 16), 256, 0, stream>>>(
      x, lengths, dec_wi_f, dec_b_f, dec_wi_b, dec_b_b, gw, gb,
      df, db, h_ws, c_ws, (float*)d_out);
}
